// CausalSelfAttention_14121852469322
// MI455X (gfx1250) — matrix operand order repaired and checked by reading, not compiled, not run on hardware
//
#include <hip/hip_runtime.h>

typedef _Float16 v16h __attribute__((ext_vector_type(16)));
typedef _Float16 v8h  __attribute__((ext_vector_type(8)));
typedef float    v8f  __attribute__((ext_vector_type(8)));
typedef unsigned v4u  __attribute__((ext_vector_type(4)));
typedef int      v8i  __attribute__((ext_vector_type(8)));
typedef int      v4i  __attribute__((ext_vector_type(4)));

#define D_MODEL  1024
#define NHEAD    16
#define HEAD_DIM 64
#define BATCH    2
#define SEQ      2048
#define M_TOTAL  (BATCH*SEQ)        // 4096

// ---------------- WMMA helper: D = A(16x32 f16) * B(32x16 f16) + C(f32) ----
__device__ __forceinline__ v8f wmma16(v16h a, v16h b, v8f c) {
  return __builtin_amdgcn_wmma_f32_16x16x32_f16(false, a, false, b, (short)0, c,
                                                false, false);
}

// Build a 16xK=32 A-fragment from two 8-half chunks (K = k0..k0+7 , k0+16..k0+23
// per lane per ISA 7.12.2 16-bit A layout). Works for global or LDS pointers.
__device__ __forceinline__ v16h load_a16(const _Float16* p0, const _Float16* p1) {
  v8h lo = *(const v8h*)p0;
  v8h hi = *(const v8h*)p1;
  v16h r;
#pragma unroll
  for (int i = 0; i < 8; ++i) { r[i] = lo[i]; r[i + 8] = hi[i]; }
  return r;
}

// ---------------- TDM: 1-D linear copy global -> LDS (n_qwords x 8 bytes) ----
__device__ __forceinline__ void tdm_load_linear(const void* gsrc,
                                                unsigned lds_byte_off,
                                                unsigned n_qwords) {
  unsigned long long ga = (unsigned long long)gsrc;
  v4u g0;
  g0[0] = 1u;                                              // count=1, user D#
  g0[1] = lds_byte_off;                                    // lds_addr [63:32]
  g0[2] = (unsigned)ga;                                    // global_addr lo
  g0[3] = ((unsigned)(ga >> 32) & 0x01ffffffu) | 0x80000000u; // addr[56:32]|type=2
  v8i g1;
  g1[0] = 0x00030000;                                      // data_size=3 (8B)
  g1[1] = (int)((n_qwords & 0xffffu) << 16);               // tensor_dim0[15:0]
  g1[2] = (int)((n_qwords >> 16) & 0xffffu) | (1 << 16);   // dim0 hi | tensor_dim1=1
  g1[3] = (int)((n_qwords & 0xffffu) << 16);               // tile_dim0
  g1[4] = 0;                                               // tile_dim1/2 = 0
  g1[5] = (int)n_qwords;                                   // tensor_dim0_stride lo
  g1[6] = 0;
  g1[7] = 0;
  v4i zz = {0, 0, 0, 0};
#if defined(__clang_major__) && (__clang_major__ >= 23)
  v8i z8 = {0, 0, 0, 0, 0, 0, 0, 0};
  __builtin_amdgcn_tensor_load_to_lds(g0, g1, zz, zz, z8, 0);
#else
  __builtin_amdgcn_tensor_load_to_lds(g0, g1, zz, zz, 0);
#endif
}

// cross-lane row reductions confined to 16-lane halves (xor masks < 16)
__device__ __forceinline__ float hmax16(float x) {
  x = fmaxf(x, __shfl_xor(x, 1));
  x = fmaxf(x, __shfl_xor(x, 2));
  x = fmaxf(x, __shfl_xor(x, 4));
  x = fmaxf(x, __shfl_xor(x, 8));
  return x;
}
__device__ __forceinline__ float hsum16(float x) {
  x += __shfl_xor(x, 1);
  x += __shfl_xor(x, 2);
  x += __shfl_xor(x, 4);
  x += __shfl_xor(x, 8);
  return x;
}

// ---------------- Kernel 0: fp32 -> f16 packing --------------------------
__global__ void __launch_bounds__(256) cvt_kernel(
    const float* __restrict__ x,
    const float* __restrict__ wq, const float* __restrict__ wk,
    const float* __restrict__ wv, const float* __restrict__ wo,
    _Float16* __restrict__ xh, _Float16* __restrict__ wh,
    _Float16* __restrict__ woh) {
  unsigned i = blockIdx.x * 256u + threadIdx.x;
  const unsigned NX = (unsigned)M_TOTAL * D_MODEL;       // 4194304
  const unsigned NW = (unsigned)D_MODEL * D_MODEL;       // 1048576
  if (i < NX) { xh[i] = (_Float16)x[i]; return; }
  unsigned j = i - NX;
  if (j < 3u * NW) {
    const float* w = (j < NW) ? wq : (j < 2u * NW) ? wk : wv;
    wh[j] = (_Float16)w[j & (NW - 1u)];
    return;
  }
  j -= 3u * NW;
  if (j < NW) woh[j] = (_Float16)wo[j];
}

// ---------------- Kernel 1/4: GEMM  Y = A @ W^T + b ----------------------
// block = 256 thr (8 waves); block tile 256(M) x 32(N); wave tile 32x32
// (2 A-frags x 2 B-frags -> 4 WMMA per k-step: halves per-WMMA operand traffic).
// Weights (32 rows x 1024 K, f16 = 64KB) staged in LDS via TDM.
// mode 0/1/2: write f16 Q/K ([b][h][s][hd]) or V transposed ([b][h][hd][s]);
// mode 3: write fp32 + bias to d_out.
__global__ void __launch_bounds__(256) gemm_kernel(
    const _Float16* __restrict__ A, const _Float16* __restrict__ W,
    const float* __restrict__ bq, const float* __restrict__ bk,
    const float* __restrict__ bv, const float* __restrict__ bo,
    _Float16* __restrict__ qh, _Float16* __restrict__ kh,
    _Float16* __restrict__ vT, float* __restrict__ outF, int mode_base) {
  __shared__ __align__(64) _Float16 wlds[32 * 1024];

  const int z = blockIdx.z;
  const int mode = mode_base + z;
  const _Float16* Wz =
      W + (size_t)z * D_MODEL * D_MODEL + (size_t)blockIdx.y * 32 * D_MODEL;

  if (threadIdx.x < 32) {                      // one wave drives the TDM
    tdm_load_linear(Wz, (unsigned)(unsigned long long)(void*)wlds,
                    32u * 1024u / 4u);         // 8192 x 8B = 64KB
    __builtin_amdgcn_s_wait_tensorcnt(0);
  }
  __syncthreads();

  const int wid = threadIdx.x >> 5;
  const int lane = threadIdx.x & 31;
  const int l16 = lane & 15;
  const int hi = lane >> 4;

  v8f cacc[2][2];                              // [m-sub][n-sub]
#pragma unroll
  for (int mi = 0; mi < 2; ++mi)
#pragma unroll
    for (int j = 0; j < 2; ++j)
#pragma unroll
      for (int e = 0; e < 8; ++e) cacc[mi][j][e] = 0.f;

  const int mrow = blockIdx.x * 256 + wid * 32;
  const _Float16* Ap0 = A + (size_t)(mrow + l16) * D_MODEL;
  const _Float16* Ap1 = Ap0 + (size_t)16 * D_MODEL;

  for (int k = 0; k < D_MODEL; k += 32) {
    v16h a0 = load_a16(Ap0 + k + hi * 8, Ap0 + k + 16 + hi * 8);
    v16h a1 = load_a16(Ap1 + k + hi * 8, Ap1 + k + 16 + hi * 8);
    v16h b0 = load_a16(wlds + ((size_t)l16 * 1024 + k + hi * 8), wlds + ((size_t)l16 * 1024 + k + 16 + hi * 8));
    v16h b1 = load_a16(wlds + ((size_t)(16 + l16) * 1024 + k + hi * 8), wlds + ((size_t)(16 + l16) * 1024 + k + 16 + hi * 8));
    cacc[0][0] = wmma16(a0, b0, cacc[0][0]);
    cacc[0][1] = wmma16(a0, b1, cacc[0][1]);
    cacc[1][0] = wmma16(a1, b0, cacc[1][0]);
    cacc[1][1] = wmma16(a1, b1, cacc[1][1]);
  }

  const float* bias = (mode == 0) ? bq : (mode == 1) ? bk : (mode == 2) ? bv : bo;
#pragma unroll
  for (int mi = 0; mi < 2; ++mi) {
#pragma unroll
    for (int e = 0; e < 8; ++e) {
      int m = mrow + mi * 16 + e + hi * 8;   // C layout: row = e + (lane>=16)*8
      int bb = m >> 11;
      int s = m & (SEQ - 1);
#pragma unroll
      for (int j = 0; j < 2; ++j) {
        int n = blockIdx.y * 32 + j * 16 + l16;
        float v = cacc[mi][j][e] + bias[n];
        int h = n >> 6, d = n & 63;
        if (mode == 0)
          qh[(((size_t)bb * NHEAD + h) * SEQ + s) * HEAD_DIM + d] = (_Float16)v;
        else if (mode == 1)
          kh[(((size_t)bb * NHEAD + h) * SEQ + s) * HEAD_DIM + d] = (_Float16)v;
        else if (mode == 2)
          vT[(((size_t)bb * NHEAD + h) * HEAD_DIM + d) * SEQ + s] = (_Float16)v;
        else
          outF[(size_t)m * D_MODEL + n] = v;
      }
    }
  }
}

// ---------------- Kernel 2: RoPE in place on Q,K --------------------------
__global__ void __launch_bounds__(256) rope_kernel(_Float16* __restrict__ qh,
                                                   _Float16* __restrict__ kh) {
  unsigned idx = blockIdx.x * 256u + threadIdx.x;   // (bh, s, j) : 32*2048*32
  if (idx >= 2097152u) return;
  const int j = idx & 31;
  const int s = (idx >> 5) & (SEQ - 1);
  const int bh = idx >> 16;
  float invf = __expf(-(float)j * 0.28782313663f);  // ln(10000)/32
  float ang = (float)s * invf;
  float c, sn;
  __sincosf(ang, &c, &sn);
  size_t base = ((size_t)bh * SEQ + s) * HEAD_DIM;
  float q1 = (float)qh[base + j], q2 = (float)qh[base + j + 32];
  qh[base + j]      = (_Float16)(q1 * c - q2 * sn);
  qh[base + j + 32] = (_Float16)(q2 * c + q1 * sn);
  float k1 = (float)kh[base + j], k2 = (float)kh[base + j + 32];
  kh[base + j]      = (_Float16)(k1 * c - k2 * sn);
  kh[base + j + 32] = (_Float16)(k2 * c + k1 * sn);
}

// ---------------- Kernel 3: causal flash attention ------------------------
// grid (S/256, B*H); block 256 = 8 waves; each wave owns 32 queries as two
// 16-row sub-tiles that SHARE the per-chunk K and V B-fragments
// (16 WMMA per 32-key chunk, same 8 operand loads -> 2x arithmetic intensity).
__global__ void __launch_bounds__(256) attn_kernel(
    const _Float16* __restrict__ qh, const _Float16* __restrict__ kh,
    const _Float16* __restrict__ vT, _Float16* __restrict__ aout) {
  __shared__ __align__(64) _Float16 pscr[8][32 * 32];  // per-wave P staging

  const int wid = threadIdx.x >> 5;
  const int lane = threadIdx.x & 31;
  const int l16 = lane & 15;
  const int hi = lane >> 4;
  const int bh = blockIdx.y;
  const int qbase = blockIdx.x * 256 + wid * 32;

  // Q as A-fragments for both 16-row sub-tiles (hd 0..31 / 32..63 each)
  const _Float16* qp0 = qh + ((size_t)bh * SEQ + qbase + l16) * HEAD_DIM;
  const _Float16* qp1 = qp0 + (size_t)16 * HEAD_DIM;
  v16h a_lo[2], a_hi[2];
  a_lo[0] = load_a16(qp0 + hi * 8, qp0 + 16 + hi * 8);
  a_hi[0] = load_a16(qp0 + 32 + hi * 8, qp0 + 48 + hi * 8);
  a_lo[1] = load_a16(qp1 + hi * 8, qp1 + 16 + hi * 8);
  a_hi[1] = load_a16(qp1 + 32 + hi * 8, qp1 + 48 + hi * 8);

  const _Float16* kp0 = kh + (size_t)bh * SEQ * HEAD_DIM;
  const _Float16* vp0 = vT + (size_t)bh * HEAD_DIM * SEQ;

  v8f o[2][4];
  float mrow[2][8], lrow[2][8];
#pragma unroll
  for (int mi = 0; mi < 2; ++mi)
#pragma unroll
    for (int e = 0; e < 8; ++e) {
      o[mi][0][e] = o[mi][1][e] = o[mi][2][e] = o[mi][3][e] = 0.f;
      mrow[mi][e] = -1e30f;
      lrow[mi][e] = 0.f;
    }

  const int nchunks = (qbase + 31) / 32 + 1;   // causal bound, 32 keys/chunk
  _Float16* ps = pscr[wid];

  for (int ch = 0; ch < nchunks; ++ch) {
    const int kc = ch * 32;

    // shared K fragments for this chunk (keys kc..kc+15 and kc+16..kc+31)
    const _Float16* kpa = kp0 + (size_t)(kc + l16) * HEAD_DIM + hi * 8;
    const _Float16* kpb = kp0 + (size_t)(kc + 16 + l16) * HEAD_DIM + hi * 8;
    v16h kA  = load_a16(kpa, kpa + 16);
    v16h kAh = load_a16(kpa + 32, kpa + 48);
    v16h kB  = load_a16(kpb, kpb + 16);
    v16h kBh = load_a16(kpb + 32, kpb + 48);

    v8f s01[2], s23[2];
#pragma unroll
    for (int mi = 0; mi < 2; ++mi) {
#pragma unroll
      for (int e = 0; e < 8; ++e) { s01[mi][e] = 0.f; s23[mi][e] = 0.f; }
      s01[mi] = wmma16(a_lo[mi], kA, s01[mi]);
      s01[mi] = wmma16(a_hi[mi], kAh, s01[mi]);
      s23[mi] = wmma16(a_lo[mi], kB, s23[mi]);
      s23[mi] = wmma16(a_hi[mi], kBh, s23[mi]);
    }

    // online softmax per sub-tile (scale 1/sqrt(64), additive causal mask)
#pragma unroll
    for (int mi = 0; mi < 2; ++mi) {
#pragma unroll
      for (int e = 0; e < 8; ++e) {
        int q = qbase + mi * 16 + e + hi * 8;
        float x0 = s01[mi][e] * 0.125f + (((kc + l16) <= q) ? 0.f : -1e9f);
        float x1 = s23[mi][e] * 0.125f + (((kc + 16 + l16) <= q) ? 0.f : -1e9f);
        float mnew = fmaxf(mrow[mi][e], hmax16(fmaxf(x0, x1)));
        float al = __expf(mrow[mi][e] - mnew);
        float p0 = __expf(x0 - mnew);
        float p1 = __expf(x1 - mnew);
        lrow[mi][e] = lrow[mi][e] * al + hsum16(p0 + p1);
        mrow[mi][e] = mnew;
        o[mi][0][e] *= al; o[mi][1][e] *= al;
        o[mi][2][e] *= al; o[mi][3][e] *= al;
        int row = mi * 16 + e + hi * 8;          // C layout -> LDS row-major
        ps[row * 32 + l16] = (_Float16)p0;
        ps[row * 32 + 16 + l16] = (_Float16)p1;
      }
    }

    // reload P as A-fragments (layout transpose through LDS)
    v16h pf0 = load_a16(ps + l16 * 32 + hi * 8, ps + l16 * 32 + 16 + hi * 8);
    v16h pf1 = load_a16(ps + (16 + l16) * 32 + hi * 8,
                        ps + (16 + l16) * 32 + 16 + hi * 8);

    // P(16x32) @ V(32x64): 4 shared V B-frags, 8 WMMAs
    const _Float16* vp = vp0 + (size_t)l16 * SEQ + kc + hi * 8;
#pragma unroll
    for (int t = 0; t < 4; ++t) {
      v16h vf = load_a16(vp + (size_t)t * 16 * SEQ, vp + (size_t)t * 16 * SEQ + 16);
      o[0][t] = wmma16(pf0, vf, o[0][t]);
      o[1][t] = wmma16(pf1, vf, o[1][t]);
    }
  }

  const int b = bh >> 4, h = bh & 15;
#pragma unroll
  for (int mi = 0; mi < 2; ++mi) {
#pragma unroll
    for (int e = 0; e < 8; ++e) {
      int q = qbase + mi * 16 + e + hi * 8;
      float inv = 1.0f / lrow[mi][e];
      size_t base =
          ((size_t)b * SEQ + q) * D_MODEL + (size_t)h * HEAD_DIM + l16;
      aout[base + 0]  = (_Float16)(o[mi][0][e] * inv);
      aout[base + 16] = (_Float16)(o[mi][1][e] * inv);
      aout[base + 32] = (_Float16)(o[mi][2][e] * inv);
      aout[base + 48] = (_Float16)(o[mi][3][e] * inv);
    }
  }
}

// ---------------- host side ------------------------------------------------
extern "C" void kernel_launch(void* const* d_in, const int* in_sizes, int n_in,
                              void* d_out, int out_size, void* d_ws,
                              size_t ws_size, hipStream_t stream) {
  (void)in_sizes; (void)n_in; (void)out_size;
  const float* x  = (const float*)d_in[0];
  // d_in[1] = attn_mask (causal mask recomputed in-kernel)
  const float* wq = (const float*)d_in[2];
  const float* bq = (const float*)d_in[3];
  const float* wk = (const float*)d_in[4];
  const float* bk = (const float*)d_in[5];
  const float* wv = (const float*)d_in[6];
  const float* bv = (const float*)d_in[7];
  const float* wo = (const float*)d_in[8];
  const float* bo = (const float*)d_in[9];
  float* out = (float*)d_out;

  char* ws = (char*)d_ws;
  size_t off = 0;
  _Float16* xh  = (_Float16*)(ws + off); off += (size_t)M_TOTAL * D_MODEL * 2;
  _Float16* wh  = (_Float16*)(ws + off); off += (size_t)3 * D_MODEL * D_MODEL * 2;
  _Float16* woh = (_Float16*)(ws + off); off += (size_t)D_MODEL * D_MODEL * 2;
  _Float16* qh  = (_Float16*)(ws + off); off += (size_t)M_TOTAL * D_MODEL * 2;
  _Float16* kh  = (_Float16*)(ws + off); off += (size_t)M_TOTAL * D_MODEL * 2;
  _Float16* vT  = (_Float16*)(ws + off); off += (size_t)M_TOTAL * D_MODEL * 2;
  _Float16* at  = (_Float16*)(ws + off); off += (size_t)M_TOTAL * D_MODEL * 2;
  if (ws_size < off) return;  // need ~48 MB of scratch

  cvt_kernel<<<32768, 256, 0, stream>>>(x, wq, wk, wv, wo, xh, wh, woh);
  gemm_kernel<<<dim3(16, 32, 3), 256, 0, stream>>>(xh, wh, bq, bk, bv, bo,
                                                   qh, kh, vT, out, 0);
  rope_kernel<<<8192, 256, 0, stream>>>(qh, kh);
  attn_kernel<<<dim3(8, 32), 256, 0, stream>>>(qh, kh, vT, at);
  gemm_kernel<<<dim3(16, 32, 1), 256, 0, stream>>>(at, woh, bq, bk, bv, bo,
                                                   qh, kh, vT, out, 3);
}
